// ClusterMultiHeadedAttention_89739046683482
// MI455X (gfx1250) — matrix operand order repaired and checked by reading, not compiled, not run on hardware
//
#include <hip/hip_runtime.h>
#include <stdint.h>

// ---------------------------------------------------------------------------
// ClusterMultiHeadedAttention for MI455X (gfx1250, wave32, WMMA, TDM)
//
//   1. proj_gemm(Wq)  : Q = Wq*x+bq   -> Qt bf16 [B][H][N][64] (transposed)
//   2. proj_gemm(Wk)  : K = Wk*x+bk   -> Kd bf16 [B][H][64][N]
//   3. proj_gemm(Wv)  : V = Wv*x+bv   -> Vd bf16 [B][H][64][N]
//   4. attention      : flash-style masked softmax (S^T = K^T Q formulation,
//                       V tiles streamed by the Tensor Data Mover, double
//                       buffered on TENSORcnt) -> nv f32 [B][256][N]
//   5. proj_gemm(Wm)  : out = Wm*nv+bm -> d_out f32
//
// Head split in the reference: c = d*NHEADS + h  (h = c&3, d = c>>2).
// ---------------------------------------------------------------------------

#define D_MODEL   256
#define N_HEADS   4
#define HEAD_DIM  64
#define BATCH     4
#define SEQ       2048

#define MODE_Q   0
#define MODE_KV  1
#define MODE_OUT 2

#define NEG_INF (-__builtin_inff())

typedef __attribute__((ext_vector_type(16))) __bf16 v16bf;
typedef __attribute__((ext_vector_type(8)))  float  v8f;

union BF16x16 { uint32_t u[8]; v16bf v; };

__device__ __forceinline__ uint16_t f32_to_bf16(float f) {
  uint32_t x = __float_as_uint(f);
  uint32_t r = x + 0x7FFFu + ((x >> 16) & 1u);   // round-to-nearest-even
  return (uint16_t)(r >> 16);
}

__device__ __forceinline__ v8f v8f_zero() {
  v8f z;
#pragma unroll
  for (int i = 0; i < 8; ++i) z[i] = 0.0f;
  return z;
}

// ---------------------------------------------------------------------------
// Tensor Data Mover: 2D tile load (tile_dim0 x tile_dim1 elements of 2 bytes)
// from a row-major tensor with row stride SEQ into packed LDS.
// This toolchain exposes the 6-arg builtin:
//   (uint32x4 g0, int32x8 g1, int32x4 g2, int32x4 g3, int32x8 pad, i32 cpol)
// ---------------------------------------------------------------------------
#if __has_builtin(__builtin_amdgcn_tensor_load_to_lds)
#define HAVE_TDM 1
typedef __attribute__((ext_vector_type(4))) unsigned int v4u_t;
typedef __attribute__((ext_vector_type(8))) int v8i_t;
typedef __attribute__((ext_vector_type(4))) int v4i_t;

__device__ __forceinline__ void tdm_load_tile32x64(uint32_t lds_off,
                                                   const uint16_t* gptr) {
  uint64_t ga = (uint64_t)(uintptr_t)gptr;          // tile start, byte address
  v4u_t g0;
  g0[0] = 1u;                                       // count=1, user descriptor
  g0[1] = lds_off;                                  // lds_addr
  g0[2] = (uint32_t)ga;                             // global_addr[31:0]
  g0[3] = (uint32_t)(ga >> 32) | (2u << 30);        // global_addr[56:32]|type=2
  v8i_t g1;
  g1[0] = 0x00010000;                               // data_size=1 (2 bytes)
  g1[1] = (int)((uint32_t)SEQ << 16);               // tensor_dim0 = SEQ
  g1[2] = (int)(64u << 16);                         // tensor_dim1 = 64
  g1[3] = (int)(32u << 16);                         // tile_dim0   = 32 (m)
  g1[4] = 64;                                       // tile_dim1   = 64 (d)
  g1[5] = SEQ;                                      // tensor_dim0_stride = SEQ
  g1[6] = 0;
  g1[7] = 0;
  v4i_t z4; z4[0] = z4[1] = z4[2] = z4[3] = 0;      // groups 2/3 unused (2D)
  v8i_t z8;
#pragma unroll
  for (int i = 0; i < 8; ++i) z8[i] = 0;
  __builtin_amdgcn_tensor_load_to_lds(g0, g1, z4, z4, z8, 0);
}
#endif

// ---------------------------------------------------------------------------
// C[256][SEQ] = W[256][256] * X[256][SEQ] + bias, per batch (blockIdx.z).
// WG = 256 threads = 8 waves; WG tile 128(M)x128(N); wave tile 64x32.
// ---------------------------------------------------------------------------
__global__ __launch_bounds__(256)
void proj_gemm_kernel(const float* __restrict__ W, const float* __restrict__ bias,
                      const float* __restrict__ X, void* __restrict__ out, int mode)
{
  __shared__ uint16_t Wl[128 * 32];   // A tile  [m][k]
  __shared__ uint16_t Xl[128 * 32];   // B tile  [n][k] (transposed)

  const int b    = blockIdx.z;
  const int m0   = blockIdx.y * 128;
  const int n0   = blockIdx.x * 128;
  const int tid  = threadIdx.x;
  const int lane = tid & 31;
  const int wave = tid >> 5;
  const int wm   = wave >> 2;
  const int wn   = wave & 3;
  const int half = lane >> 4;
  const int l16  = lane & 15;

  const float* Xb = X + (size_t)b * D_MODEL * SEQ;

  v8f acc[4][2];
#pragma unroll
  for (int i = 0; i < 4; ++i)
#pragma unroll
    for (int j = 0; j < 2; ++j) acc[i][j] = v8f_zero();

  for (int k0 = 0; k0 < D_MODEL; k0 += 32) {
    __syncthreads();
#pragma unroll
    for (int i = 0; i < 16; ++i) {                 // W tile: 128x32
      int q = tid + i * 256;
      int m = q >> 5, k = q & 31;
      Wl[q] = f32_to_bf16(W[(size_t)(m0 + m) * D_MODEL + k0 + k]);
    }
#pragma unroll
    for (int i = 0; i < 16; ++i) {                 // X tile: 32x128 -> [n][k]
      int q = tid + i * 256;
      int k = q >> 7, n = q & 127;
      Xl[n * 32 + k] = f32_to_bf16(Xb[(size_t)(k0 + k) * SEQ + n0 + n]);
    }
    __syncthreads();

    BF16x16 bb[2];
#pragma unroll
    for (int ns = 0; ns < 2; ++ns) {
      int n = wn * 32 + ns * 16 + l16;
      const uint32_t* p = (const uint32_t*)&Xl[n * 32];
#pragma unroll
      for (int v = 0; v < 8; ++v) bb[ns].u[v] = p[half * 8 + v];
    }
#pragma unroll
    for (int ms = 0; ms < 4; ++ms) {
      int m = wm * 64 + ms * 16 + l16;
      BF16x16 aa;
      const uint32_t* p = (const uint32_t*)&Wl[m * 32];
#pragma unroll
      for (int v = 0; v < 8; ++v)
        aa.u[v] = p[half * 8 + v];
#pragma unroll
      for (int ns = 0; ns < 2; ++ns) {
        acc[ms][ns] = __builtin_amdgcn_wmma_f32_16x16x32_bf16(
            false, aa.v, false, bb[ns].v, (short)0, acc[ms][ns], false, false);
      }
    }
  }

#pragma unroll
  for (int ms = 0; ms < 4; ++ms) {
#pragma unroll
    for (int ns = 0; ns < 2; ++ns) {
      int ncol = n0 + wn * 32 + ns * 16 + l16;
#pragma unroll
      for (int r = 0; r < 8; ++r) {
        int mrow = m0 + wm * 64 + ms * 16 + half * 8 + r;
        float val = acc[ms][ns][r] + bias[mrow];
        int h = mrow & 3, d = mrow >> 2;
        if (mode == MODE_Q) {
          uint16_t* o = (uint16_t*)out;
          o[((size_t)(b * N_HEADS + h) * SEQ + ncol) * HEAD_DIM + d] = f32_to_bf16(val);
        } else if (mode == MODE_KV) {
          uint16_t* o = (uint16_t*)out;
          o[((size_t)(b * N_HEADS + h) * HEAD_DIM + d) * SEQ + ncol] = f32_to_bf16(val);
        } else {
          float* o = (float*)out;
          o[((size_t)b * D_MODEL + mrow) * SEQ + ncol] = val;
        }
      }
    }
  }
}

// ---------------------------------------------------------------------------
// Flash attention, S^T = K^T Q formulation: C/D column = query = lane, so the
// masked-softmax row reduction is 16 in-lane values + ONE shfl_xor(16), and
// running max/sum are single scalars per lane.
// ---------------------------------------------------------------------------
__global__ __launch_bounds__(256)
void attention_kernel(const uint16_t* __restrict__ Qt,   // [B][H][N][64]
                      const uint16_t* __restrict__ Kd,   // [B][H][64][N]
                      const uint16_t* __restrict__ Vd,   // [B][H][64][N]
                      const int* __restrict__ qlab,
                      const int* __restrict__ vlab,
                      float* __restrict__ nv)            // [B][256][N]
{
  __shared__ uint16_t Kl[2][32 * 64];     // [m][d]  transposed (A of S^T)
  __shared__ uint16_t Vl[2][64 * 32];     // [d][m]  direct     (A of O^T)
  __shared__ int      vls[2][32];
  __shared__ uint16_t Pl[8][16 * 32];     // per-wave P^T relayout [n][m]

  const int b    = blockIdx.z;
  const int h    = blockIdx.y;
  const int tid  = threadIdx.x;
  const int lane = tid & 31;
  const int wave = tid >> 5;
  const int half = lane >> 4;
  const int l16  = lane & 15;
  const int nbase = blockIdx.x * 128 + wave * 16;

  const size_t bh = (size_t)(b * N_HEADS + h);
  const uint16_t* Qbh = Qt + bh * SEQ * HEAD_DIM;
  const uint16_t* Kbh = Kd + bh * HEAD_DIM * SEQ;
  const uint16_t* Vbh = Vd + bh * HEAD_DIM * SEQ;
  const int* ql  = qlab + (size_t)b * SEQ;
  const int* vlg = vlab + (size_t)b * SEQ;

  // Q as B operand (col = query = l16), two d-halves, contiguous b32 pairs
  BF16x16 bq0, bq1;
  {
    const uint32_t* qrow = (const uint32_t*)&Qbh[(size_t)(nbase + l16) * HEAD_DIM];
#pragma unroll
    for (int v = 0; v < 8; ++v) {
      bq0.u[v] = qrow[half * 8 + v];
      bq1.u[v] = qrow[16 + half * 8 + v];
    }
  }
  const int qn = ql[nbase + l16];              // this lane's query label

  float Mrow = NEG_INF, Lrow = 0.0f;
  v8f acc[4];                                  // O^T: row d = c*16+half*8+r, col n = l16
#pragma unroll
  for (int c = 0; c < 4; ++c) acc[c] = v8f_zero();

  const float scale = 0.125f;                  // 1/sqrt(64)
  const int NT = SEQ / 32;

  // --- staging helpers -----------------------------------------------------
  // K tile: one b128 global load per thread, 8 b16 transposed LDS stores.
  auto stage_k = [&](int buf, int m0) {
    int d = tid >> 2, mc = tid & 3;
    uint4 kv = *(const uint4*)&Kbh[(size_t)d * SEQ + m0 + mc * 8];
    uint32_t w[4] = {kv.x, kv.y, kv.z, kv.w};
#pragma unroll
    for (int j = 0; j < 8; ++j) {
      uint16_t e = (uint16_t)((j & 1) ? (w[j >> 1] >> 16) : (w[j >> 1] & 0xFFFFu));
      Kl[buf][(mc * 8 + j) * 64 + d] = e;
    }
    if (tid < 32) vls[buf][tid] = vlg[m0 + tid];
  };
#ifndef HAVE_TDM
  auto stage_v = [&](int buf, int m0) {
    int d = tid >> 2, mc = tid & 3;
    *(uint4*)&Vl[buf][d * 32 + mc * 8] =
        *(const uint4*)&Vbh[(size_t)d * SEQ + m0 + mc * 8];
  };
#endif

  // --- prologue: stage tile 0 ----------------------------------------------
  stage_k(0, 0);
#ifdef HAVE_TDM
  if (wave == 0) {
    tdm_load_tile32x64((uint32_t)(size_t)&Vl[0][0], &Vbh[0]);
    __builtin_amdgcn_s_wait_tensorcnt(0);
  }
#else
  stage_v(0, 0);
#endif
  __syncthreads();

  for (int it = 0; it < NT; ++it) {
    const int cur = it & 1;
    const int m0  = it * 32;

    // prefetch next tile into the other buffer (overlaps with compute; the
    // TDM transfer is tracked by TENSORcnt, untouched by mid-loop DS waits)
    if (it + 1 < NT) {
      stage_k(1 - cur, m0 + 32);
#ifdef HAVE_TDM
      if (wave == 0) tdm_load_tile32x64((uint32_t)(size_t)&Vl[1 - cur][0],
                                        &Vbh[m0 + 32]);
#else
      stage_v(1 - cur, m0 + 32);
#endif
      __builtin_prefetch(&Kbh[(size_t)(tid >> 2) * SEQ + m0 + 64], 0, 1);
    }

    // S^T[32m x 16n] = K^T Q : 4 WMMAs
    v8f s[2];
#pragma unroll
    for (int sub = 0; sub < 2; ++sub) {
      BF16x16 a0, a1;
      const uint32_t* kr = (const uint32_t*)&Kl[cur][(sub * 16 + l16) * 64];
#pragma unroll
      for (int v = 0; v < 8; ++v) {
        int idx = half * 8 + v;
        a0.u[v] = kr[idx];
        a1.u[v] = kr[idx + 16];
      }
      v8f z = v8f_zero();
      z = __builtin_amdgcn_wmma_f32_16x16x32_bf16(false, a0.v, false, bq0.v,
                                                  (short)0, z, false, false);
      s[sub] = __builtin_amdgcn_wmma_f32_16x16x32_bf16(false, a1.v, false, bq1.v,
                                                       (short)0, z, false, false);
    }

    // online masked softmax: 16 keys/lane, combine halves with one shuffle
    const int* labc = &vls[cur][half * 8];     // m = sub*16 + half*8 + r
    float mx = NEG_INF;
#pragma unroll
    for (int sub = 0; sub < 2; ++sub)
#pragma unroll
      for (int r = 0; r < 8; ++r)
        if (labc[sub * 16 + r] == qn) mx = fmaxf(mx, s[sub][r] * scale);
    mx = fmaxf(mx, __shfl_xor(mx, 16, 32));
    const float Mn    = fmaxf(Mrow, mx);
    const float alpha = (Mrow == NEG_INF) ? 0.0f : __expf(Mrow - Mn);

    float rs = 0.0f;
    uint32_t* prow = (uint32_t*)&Pl[wave][l16 * 32];
#pragma unroll
    for (int sub = 0; sub < 2; ++sub)
#pragma unroll
      for (int rp = 0; rp < 4; ++rp) {
        int r0 = rp * 2, r1 = rp * 2 + 1;
        bool mk0 = labc[sub * 16 + r0] == qn;
        bool mk1 = labc[sub * 16 + r1] == qn;
        float p0 = mk0 ? __expf(s[sub][r0] * scale - Mn) : 0.0f;
        float p1 = mk1 ? __expf(s[sub][r1] * scale - Mn) : 0.0f;
        rs += p0 + p1;
        prow[sub * 8 + half * 4 + rp] =
            (uint32_t)f32_to_bf16(p0) | ((uint32_t)f32_to_bf16(p1) << 16);
      }
    rs += __shfl_xor(rs, 16, 32);
    Lrow = Lrow * alpha + rs;
    Mrow = Mn;

#pragma unroll
    for (int c = 0; c < 4; ++c)
#pragma unroll
      for (int r = 0; r < 8; ++r) acc[c][r] *= alpha;

    // wave-private LDS RAW on P^T relayout
    asm volatile("s_wait_dscnt 0" ::: "memory");

    BF16x16 pb;                                // P^T as B operand (col = n)
    {
      const uint32_t* pr = (const uint32_t*)&Pl[wave][l16 * 32];
#pragma unroll
      for (int v = 0; v < 8; ++v) pb.u[v] = pr[half * 8 + v];
    }
    // O^T[64d x 16n] += V[16d x 32m] * P^T[32m x 16n] : 4 WMMAs
#pragma unroll
    for (int c = 0; c < 4; ++c) {
      BF16x16 va;
      const uint32_t* vr = (const uint32_t*)&Vl[cur][(c * 16 + l16) * 32];
#pragma unroll
      for (int v = 0; v < 8; ++v)
        va.u[v] = vr[half * 8 + v];
      acc[c] = __builtin_amdgcn_wmma_f32_16x16x32_bf16(
          false, va.v, false, pb.v, (short)0, acc[c], false, false);
    }

#ifdef HAVE_TDM
    if (wave == 0) __builtin_amdgcn_s_wait_tensorcnt(0);
#endif
    __syncthreads();
  }

  // epilogue: one 1/L per lane (rows with no matching label keep L==0 -> 0)
  const float inv = (Lrow > 0.0f) ? (1.0f / Lrow) : 0.0f;
  const int n = nbase + l16;
#pragma unroll
  for (int c = 0; c < 4; ++c)
#pragma unroll
    for (int r = 0; r < 8; ++r) {
      int d  = c * 16 + half * 8 + r;
      int ch = d * N_HEADS + h;                // reference head interleave
      nv[((size_t)b * D_MODEL + ch) * SEQ + n] = acc[c][r] * inv;
    }
}

// ---------------------------------------------------------------------------
extern "C" void kernel_launch(void* const* d_in, const int* in_sizes, int n_in,
                              void* d_out, int out_size, void* d_ws, size_t ws_size,
                              hipStream_t stream) {
  (void)in_sizes; (void)n_in; (void)out_size; (void)ws_size;
  const float* query = (const float*)d_in[0];
  const float* key   = (const float*)d_in[1];
  const float* value = (const float*)d_in[2];
  const int*   qlab  = (const int*)d_in[3];
  const int*   vlab  = (const int*)d_in[4];
  const float* Wq = (const float*)d_in[5];  const float* bq = (const float*)d_in[6];
  const float* Wk = (const float*)d_in[7];  const float* bk = (const float*)d_in[8];
  const float* Wv = (const float*)d_in[9];  const float* bv = (const float*)d_in[10];
  const float* Wm = (const float*)d_in[11]; const float* bm = (const float*)d_in[12];

  // workspace: Qt/Kd/Vd bf16 (4 MB each) + nv f32 (8 MB) = 20 MB
  const size_t qkv_elems = (size_t)BATCH * N_HEADS * SEQ * HEAD_DIM;
  uint16_t* Qt = (uint16_t*)d_ws;
  uint16_t* Kd = Qt + qkv_elems;
  uint16_t* Vd = Kd + qkv_elems;
  float*    nv = (float*)(Vd + qkv_elems);

  dim3 gGemm(SEQ / 128, D_MODEL / 128, BATCH);   // 16 x 2 x 4
  dim3 gAttn(SEQ / 128, N_HEADS, BATCH);         // 16 x 4 x 4
  dim3 blk(256);

  proj_gemm_kernel<<<gGemm, blk, 0, stream>>>(Wq, bq, query, (void*)Qt, MODE_Q);
  proj_gemm_kernel<<<gGemm, blk, 0, stream>>>(Wk, bk, key,   (void*)Kd, MODE_KV);
  proj_gemm_kernel<<<gGemm, blk, 0, stream>>>(Wv, bv, value, (void*)Vd, MODE_KV);
  attention_kernel<<<gAttn, blk, 0, stream>>>(Qt, Kd, Vd, qlab, vlab, nv);
  proj_gemm_kernel<<<gGemm, blk, 0, stream>>>(Wm, bm, nv, d_out, MODE_OUT);
}
